// MultiHeadCrossModalAttention_21732534518179
// MI455X (gfx1250) — matrix operand order repaired and checked by reading, not compiled, not run on hardware
//
#include <hip/hip_runtime.h>
#include <hip/hip_bf16.h>

typedef _Float16 h16;
typedef __attribute__((ext_vector_type(8)))  _Float16 v8h;
typedef __attribute__((ext_vector_type(16))) _Float16 v16h;
typedef __attribute__((ext_vector_type(8)))  float    v8f;
typedef __attribute__((ext_vector_type(4)))  unsigned int v4u;
typedef __attribute__((ext_vector_type(8)))  int      v8i;
typedef __attribute__((ext_vector_type(4)))  int      v4i;

#define E_DIM 1024
#define NHEAD 16
#define HDIM  64
#define SQ_LEN 1024
#define SK_LEN 2048
#define BATCH 4
#define BK 32

#if __has_builtin(__builtin_amdgcn_tensor_load_to_lds)
#define HAS_TDM 1
#else
#define HAS_TDM 0
#endif

// ---------------------------------------------------------------------------
// CDNA5 helpers
// ---------------------------------------------------------------------------

// Generic->LDS address: compiler's own flat->local addrspacecast is a
// truncation to the low 32 bits, which is what VDST of async/DS ops wants.
__device__ __forceinline__ unsigned lds_addr32(const void* p) {
  return (unsigned)(unsigned long long)p;
}

// Async byte copy global->LDS (16B per lane), tracked by ASYNCcnt.
__device__ __forceinline__ void async_copy_b128(const void* g, void* lds) {
  unsigned l = lds_addr32(lds);
  unsigned long long ga = (unsigned long long)g;
  asm volatile("global_load_async_to_lds_b128 %0, %1, off"
               :: "v"(l), "v"(ga) : "memory");
}
__device__ __forceinline__ void wait_async0() {
  asm volatile("s_wait_asynccnt 0" ::: "memory");
}

#if HAS_TDM
// 2D TDM tensor tile load: tile (tile_d0 x tile_d1) of 2-byte elements from a
// row-major tensor with row stride stride0, packed contiguously into LDS.
__device__ __forceinline__ void tdm_load_2d_f16(const void* gaddr, void* ldsaddr,
                                                unsigned tensor_d0, unsigned tensor_d1,
                                                unsigned long long stride0,
                                                unsigned tile_d0, unsigned tile_d1) {
  unsigned long long ga = (unsigned long long)gaddr;
  unsigned la = lds_addr32(ldsaddr);
  v4u g0;
  g0[0] = 1u;                                   // count=1 (valid user descriptor)
  g0[1] = la;                                   // lds_addr
  g0[2] = (unsigned)ga;                         // global_addr[31:0]
  g0[3] = (unsigned)(ga >> 32) | (2u << 30);    // global_addr[56:32] | type=2
  v8i g1;
  g1[0] = (int)(1u << 16);                      // data_size=1 (2 bytes), no mask/pad
  g1[1] = (int)((tensor_d0 & 0xFFFFu) << 16);   // tensor_dim0[15:0] @ bits 63:48
  g1[2] = (int)(((tensor_d0 >> 16) & 0xFFFFu) | ((tensor_d1 & 0xFFFFu) << 16));
  g1[3] = (int)(((tensor_d1 >> 16) & 0xFFFFu) | (tile_d0 << 16));
  g1[4] = (int)(tile_d1 & 0xFFFFu);             // tile_dim1 | tile_dim2=0
  g1[5] = (int)(unsigned)stride0;               // tensor_dim0_stride[31:0]
  g1[6] = (int)((stride0 >> 32) & 0xFFFFu);     // stride[47:32] | dim1_stride lo = 0
  g1[7] = 0;
  v4i gz4 = {0, 0, 0, 0};                       // groups 2/3 unused (2D tile)
  v8i gz8 = {0, 0, 0, 0, 0, 0, 0, 0};           // extended group (clang-23 6-arg form)
  __builtin_amdgcn_tensor_load_to_lds(g0, g1, gz4, gz4, gz8, 0);
}
#endif

// Build a 16-half WMMA fragment from two contiguous 8-half chunks.
__device__ __forceinline__ v16h make_frag(const h16* lo, const h16* hi) {
  v8h a = *(const v8h*)lo;
  v8h b = *(const v8h*)hi;
  v16h r;
#pragma unroll
  for (int i = 0; i < 8; ++i) { r[i] = a[i]; r[i + 8] = b[i]; }
  return r;
}

__device__ __forceinline__ float redmax16(float x) {
#pragma unroll
  for (int m = 1; m < 16; m <<= 1) x = fmaxf(x, __shfl_xor(x, m, 32));
  return x;
}
__device__ __forceinline__ float redsum16(float x) {
#pragma unroll
  for (int m = 1; m < 16; m <<= 1) x += __shfl_xor(x, m, 32);
  return x;
}

// ---------------------------------------------------------------------------
// fp32 -> f16 elementwise (weights pre-conversion)
// ---------------------------------------------------------------------------
__global__ __launch_bounds__(256) void f32_to_f16_kernel(const float* __restrict__ x,
                                                         h16* __restrict__ y) {
  int i = (blockIdx.x * 256 + threadIdx.x) * 4;
#pragma unroll
  for (int j = 0; j < 4; ++j) y[i + j] = (h16)x[i + j];
}

// ---------------------------------------------------------------------------
// C[M,E] = A[M,E] @ W[E,E]^T + bias  (nn.Linear), W already f16.
// A_F16: A is f16 (ctx buffer) -> staged via TDM; else fp32, converted in LDS.
// OUT_F32RES: fp32 out + residual (final projection); else f16 out.
// Block tile 128x64, 8 waves, each wave 2x2 WMMA 16x16x32 tiles.
// ---------------------------------------------------------------------------
template <bool A_F16, bool OUT_F32RES>
__global__ __launch_bounds__(256) void gemm_bias_kernel(
    const void* __restrict__ Ap, const h16* __restrict__ Wh,
    const float* __restrict__ bias, const float* __restrict__ residual,
    void* __restrict__ outp) {
  __shared__ __align__(16) h16 As[128 * BK];
  __shared__ __align__(16) h16 Ws[64 * BK];

  const int t = threadIdx.x;
  const int wave = t >> 5;
  const int lane = t & 31;
  const int nlo = lane & 15;
  const int sel = lane >> 4;
  const int m_blk = blockIdx.y * 128;
  const int n_blk = blockIdx.x * 64;
  const int wm = (wave >> 1) * 32;
  const int wn = (wave & 1) * 32;

  const float* Af = (const float*)Ap;
  const h16* Ah = (const h16*)Ap;

  v8f acc[2][2] = {};

  for (int kk = 0; kk < E_DIM; kk += BK) {
    __syncthreads();  // previous tiles fully consumed
#if HAS_TDM
    if (t < 32) {  // wave 0 drives the Tensor Data Mover
      tdm_load_2d_f16(Wh + (size_t)n_blk * E_DIM + kk, Ws,
                      E_DIM, 64, E_DIM, BK, 64);
      if (A_F16)
        tdm_load_2d_f16(Ah + (size_t)m_blk * E_DIM + kk, As,
                        E_DIM, 128, E_DIM, BK, 128);
      __builtin_amdgcn_s_wait_tensorcnt(0);
    }
#else
    {  // fallback: cooperative f16 tile copies
      int r = t >> 2, c8 = (t & 3) * 8;
      *(v8h*)&Ws[r * BK + c8] = *(const v8h*)&Wh[(size_t)(n_blk + r) * E_DIM + kk + c8];
      if (A_F16) {
#pragma unroll
        for (int i = 0; i < 2; ++i) {
          int rr = r + 64 * i;
          *(v8h*)&As[rr * BK + c8] =
              *(const v8h*)&Ah[(size_t)(m_blk + rr) * E_DIM + kk + c8];
        }
      }
    }
#endif
    if (!A_F16) {  // fp32 A: load + convert while staging
#pragma unroll
      for (int i = 0; i < 16; ++i) {
        int idx = i * 256 + t;
        int r = idx >> 5, c = idx & 31;
        As[r * BK + c] = (h16)Af[(size_t)(m_blk + r) * E_DIM + kk + c];
      }
    }
    __syncthreads();

    v16h afr[2], bfr[2];
#pragma unroll
    for (int mi = 0; mi < 2; ++mi) {
      // A 16x32 layout: lane=row; halves [8*sel, +8) and [16+8*sel, +8)
      const h16* base = &As[(wm + mi * 16 + nlo) * BK];
      afr[mi] = make_frag(base + 8 * sel, base + 16 + 8 * sel);
    }
#pragma unroll
    for (int ni = 0; ni < 2; ++ni) {
      // B 32x16 layout: lane=col; halves [8*sel, +8) and [16+8*sel, +8)
      const h16* base = &Ws[(wn + ni * 16 + nlo) * BK];
      bfr[ni] = make_frag(base + 8 * sel, base + 16 + 8 * sel);
    }
#pragma unroll
    for (int mi = 0; mi < 2; ++mi)
#pragma unroll
      for (int ni = 0; ni < 2; ++ni)
        acc[mi][ni] = __builtin_amdgcn_wmma_f32_16x16x32_f16(
            false, afr[mi], false, bfr[ni], (short)0, acc[mi][ni], false, false);
  }

  // epilogue: C layout col=lane&15, rows v + 8*sel
#pragma unroll
  for (int mi = 0; mi < 2; ++mi) {
#pragma unroll
    for (int ni = 0; ni < 2; ++ni) {
      int col = n_blk + wn + ni * 16 + nlo;
      float bv = bias[col];
#pragma unroll
      for (int v = 0; v < 8; ++v) {
        int row = m_blk + wm + mi * 16 + v + 8 * sel;
        float val = acc[mi][ni][v] + bv;
        if (OUT_F32RES) {
          ((float*)outp)[(size_t)row * E_DIM + col] =
              val + residual[(size_t)row * E_DIM + col];
        } else {
          ((h16*)outp)[(size_t)row * E_DIM + col] = (h16)val;
        }
      }
    }
  }
}

// ---------------------------------------------------------------------------
// Flash attention: grid (SQ/128, B*H), 8 waves x 16 query rows.
// 64-key steps: K tile async-copied to LDS (ASYNCcnt), V tile transposed via
// LDS, 8 score WMMAs + online softmax + 8 ctx WMMAs per step.
// ---------------------------------------------------------------------------
__global__ __launch_bounds__(256) void attention_kernel(
    const h16* __restrict__ Q, const h16* __restrict__ Kb,
    const h16* __restrict__ Vb, h16* __restrict__ Ctx) {
  __shared__ __align__(16) h16 Ks[64][64];       // [key][d]
  __shared__ __align__(16) h16 Vt[64][64];       // [d][key] (transposed)
  __shared__ __align__(16) h16 Plds[8][16][64];  // per-wave P bounce

  const int t = threadIdx.x;
  const int wave = t >> 5;
  const int lane = t & 31;
  const int nlo = lane & 15;
  const int sel = lane >> 4;
  const int b = blockIdx.y >> 4;
  const int h = blockIdx.y & 15;
  const int q0 = blockIdx.x * 128 + wave * 16;

  // Q fragments (head-dim chunks 0..31, 32..63) kept in VGPRs
  v16h qa[2];
  {
    const h16* qrow = Q + (size_t)(b * SQ_LEN + q0 + nlo) * E_DIM + h * HDIM;
#pragma unroll
    for (int tt = 0; tt < 2; ++tt)
      qa[tt] = make_frag(qrow + 32 * tt + 8 * sel, qrow + 32 * tt + 16 + 8 * sel);
  }

  float mrow[8], lrow[8];
  v8f acc[4] = {};
#pragma unroll
  for (int v = 0; v < 8; ++v) { mrow[v] = -1e30f; lrow[v] = 0.0f; }

  const int krow = t >> 3, kchunk = (t & 7) * 8;  // async K copy mapping
  const int vrow = t >> 2, vseg = (t & 3) * 16;   // V transpose mapping

  for (int kk = 0; kk < SK_LEN; kk += 64) {
    __syncthreads();  // previous K/V tiles fully consumed

    // K tile (64x64 f16 = 8KB): pure async byte copy, 2x16B per thread
#pragma unroll
    for (int half = 0; half < 2; ++half) {
      int r = krow + 32 * half;
      async_copy_b128(Kb + (size_t)(b * SK_LEN + kk + r) * E_DIM + h * HDIM + kchunk,
                      &Ks[r][kchunk]);
    }
    // V tile transposed into LDS (needs element shuffle -> manual path)
    {
      const h16* src = Vb + (size_t)(b * SK_LEN + kk + vrow) * E_DIM + h * HDIM + vseg;
      v8h a = *(const v8h*)src;
      v8h c = *(const v8h*)(src + 8);
#pragma unroll
      for (int j = 0; j < 8; ++j) {
        Vt[vseg + j][vrow] = a[j];
        Vt[vseg + 8 + j][vrow] = c[j];
      }
    }
    if (kk + 64 < SK_LEN) {  // prefetch next tiles into cache
      __builtin_prefetch(Kb + (size_t)(b * SK_LEN + kk + 64 + krow) * E_DIM + h * HDIM, 0, 1);
      __builtin_prefetch(Vb + (size_t)(b * SK_LEN + kk + 64 + krow) * E_DIM + h * HDIM, 0, 1);
    }
    wait_async0();
    __syncthreads();  // K/V tiles ready

    // scores: four 16-key tiles, contraction over D=64 (2 WMMAs each)
    v8f s[4];
#pragma unroll
    for (int st = 0; st < 4; ++st) {
      v8f c = {};
#pragma unroll
      for (int tt = 0; tt < 2; ++tt) {
        const h16* base = &Ks[st * 16 + nlo][32 * tt + 8 * sel];
        v16h kf = make_frag(base, base + 16);
        c = __builtin_amdgcn_wmma_f32_16x16x32_f16(false, qa[tt], false, kf,
                                                   (short)0, c, false, false);
      }
#pragma unroll
      for (int v = 0; v < 8; ++v) c[v] *= 0.125f;  // 1/sqrt(64)
      s[st] = c;
    }

    // online softmax across 64 keys (rows = v + 8*sel)
#pragma unroll
    for (int v = 0; v < 8; ++v) {
      float tmax = fmaxf(fmaxf(s[0][v], s[1][v]), fmaxf(s[2][v], s[3][v]));
      tmax = redmax16(tmax);
      float mnew = fmaxf(mrow[v], tmax);
      float corr = __expf(mrow[v] - mnew);
#pragma unroll
      for (int st = 0; st < 4; ++st) s[st][v] = __expf(s[st][v] - mnew);
      float rs = redsum16(s[0][v] + s[1][v] + s[2][v] + s[3][v]);
      lrow[v] = lrow[v] * corr + rs;
      mrow[v] = mnew;
#pragma unroll
      for (int ni = 0; ni < 4; ++ni) acc[ni][v] *= corr;
    }

    // P: C-layout -> LDS -> two A-layout fragments (same-wave DS, in order)
#pragma unroll
    for (int st = 0; st < 4; ++st)
#pragma unroll
      for (int v = 0; v < 8; ++v)
        Plds[wave][v + 8 * sel][st * 16 + nlo] = (h16)s[st][v];

    const h16* pb = &Plds[wave][nlo][0];
    v16h pf[2];
#pragma unroll
    for (int c = 0; c < 2; ++c)
      pf[c] = make_frag(pb + 32 * c + 8 * sel, pb + 32 * c + 16 + 8 * sel);

    // ctx += P(16x64) x V(64x64): 4 N-tiles x 2 K-chunks
#pragma unroll
    for (int ni = 0; ni < 4; ++ni) {
#pragma unroll
      for (int c = 0; c < 2; ++c) {
        const h16* vbp = &Vt[ni * 16 + nlo][32 * c + 8 * sel];
        v16h vf = make_frag(vbp, vbp + 16);
        acc[ni] = __builtin_amdgcn_wmma_f32_16x16x32_f16(false, pf[c], false, vf,
                                                         (short)0, acc[ni], false, false);
      }
    }
  }

  // normalize and store ctx (f16, [B*SQ][E], col = h*64+d)
#pragma unroll
  for (int v = 0; v < 8; ++v) {
    float rinv = 1.0f / lrow[v];
    int row = b * SQ_LEN + q0 + v + 8 * sel;
#pragma unroll
    for (int ni = 0; ni < 4; ++ni)
      Ctx[(size_t)row * E_DIM + h * HDIM + ni * 16 + nlo] = (h16)(acc[ni][v] * rinv);
  }
}

// ---------------------------------------------------------------------------
// Row LayerNorm: one block per row of 1024, 4 elements per thread.
// ---------------------------------------------------------------------------
__global__ __launch_bounds__(256) void layernorm_kernel(
    const float* __restrict__ X, const float* __restrict__ gamma,
    const float* __restrict__ beta, float* __restrict__ out) {
  __shared__ float red[16];
  const int row = blockIdx.x;
  const int t = threadIdx.x;
  const float* xr = X + (size_t)row * E_DIM;

  float x[4], s = 0.0f, s2 = 0.0f;
#pragma unroll
  for (int j = 0; j < 4; ++j) {
    x[j] = xr[t + 256 * j];
    s += x[j];
    s2 += x[j] * x[j];
  }
#pragma unroll
  for (int m = 1; m < 32; m <<= 1) {
    s += __shfl_xor(s, m, 32);
    s2 += __shfl_xor(s2, m, 32);
  }
  if ((t & 31) == 0) { red[t >> 5] = s; red[8 + (t >> 5)] = s2; }
  __syncthreads();
  s = 0.0f; s2 = 0.0f;
#pragma unroll
  for (int w = 0; w < 8; ++w) { s += red[w]; s2 += red[8 + w]; }

  float mu = s * (1.0f / E_DIM);
  float var = s2 * (1.0f / E_DIM) - mu * mu;
  float rstd = rsqrtf(var + 1e-5f);
#pragma unroll
  for (int j = 0; j < 4; ++j) {
    int c = t + 256 * j;
    out[(size_t)row * E_DIM + c] = (x[j] - mu) * rstd * gamma[c] + beta[c];
  }
}

// ---------------------------------------------------------------------------
extern "C" void kernel_launch(void* const* d_in, const int* in_sizes, int n_in,
                              void* d_out, int out_size, void* d_ws, size_t ws_size,
                              hipStream_t stream) {
  const float* query = (const float*)d_in[0];
  const float* key   = (const float*)d_in[1];
  const float* value = (const float*)d_in[2];
  const float* Wq = (const float*)d_in[3];
  const float* bq = (const float*)d_in[4];
  const float* Wk = (const float*)d_in[5];
  const float* bk = (const float*)d_in[6];
  const float* Wv = (const float*)d_in[7];
  const float* bv = (const float*)d_in[8];
  const float* Wo = (const float*)d_in[9];
  const float* bo = (const float*)d_in[10];
  const float* gamma = (const float*)d_in[11];
  const float* beta  = (const float*)d_in[12];

  char* ws = (char*)d_ws;
  const size_t MB = 1 << 20;
  h16* Qf = (h16*)(ws);                 //  8 MB projected Q (f16)
  h16* Kf = (h16*)(ws + 8 * MB);        // 16 MB projected K
  h16* Vf = (h16*)(ws + 24 * MB);       // 16 MB projected V
  h16* Cx = (h16*)(ws + 40 * MB);       //  8 MB attention context
  float* Pre = (float*)(ws + 48 * MB);  // 16 MB pre-LayerNorm (fp32)
  h16* Wqh = (h16*)(ws + 64 * MB);      //  2 MB each: f16 weights
  h16* Wkh = (h16*)(ws + 66 * MB);
  h16* Wvh = (h16*)(ws + 68 * MB);
  h16* Woh = (h16*)(ws + 70 * MB);

  dim3 blk(256);
  // weights fp32 -> f16 (1M elements each, 4/thread)
  f32_to_f16_kernel<<<dim3(1024), blk, 0, stream>>>(Wq, Wqh);
  f32_to_f16_kernel<<<dim3(1024), blk, 0, stream>>>(Wk, Wkh);
  f32_to_f16_kernel<<<dim3(1024), blk, 0, stream>>>(Wv, Wvh);
  f32_to_f16_kernel<<<dim3(1024), blk, 0, stream>>>(Wo, Woh);
  // projections: Q (M=4096), K,V (M=8192), f16 out
  gemm_bias_kernel<false, false><<<dim3(16, 32), blk, 0, stream>>>(query, Wqh, bq, nullptr, Qf);
  gemm_bias_kernel<false, false><<<dim3(16, 64), blk, 0, stream>>>(key,   Wkh, bk, nullptr, Kf);
  gemm_bias_kernel<false, false><<<dim3(16, 64), blk, 0, stream>>>(value, Wvh, bv, nullptr, Vf);
  // flash attention over 64 (b,h) pairs, 8 query blocks of 128
  attention_kernel<<<dim3(8, 64), blk, 0, stream>>>(Qf, Kf, Vf, Cx);
  // output projection + bias + residual (fp32)
  gemm_bias_kernel<true, true><<<dim3(16, 32), blk, 0, stream>>>(Cx, Woh, bo, query, Pre);
  // layernorm -> final output
  layernorm_kernel<<<dim3(4096), blk, 0, stream>>>(Pre, gamma, beta, (float*)d_out);
}
